// SAEncoder_10161892622813
// MI455X (gfx1250) — matrix operand order repaired and checked by reading, not compiled, not run on hardware
//
#include <hip/hip_runtime.h>

// ---------------------------------------------------------------------------
// Problem constants (match reference)
// ---------------------------------------------------------------------------
#define B_    32
#define N_    256
#define EMB_  256
#define H_    16
#define DQK_  16
#define FF_   512
#define L_    5
#define MTOT  (B_ * N_)          // 8192 rows for all row-major GEMMs

typedef __bf16 bf16;
typedef __attribute__((ext_vector_type(16))) __bf16 v16bf;
typedef __attribute__((ext_vector_type(8)))  float  v8f;

union Frag16 {            // one WMMA 16-bit A/B operand (8 VGPRs / lane)
    v16bf v;
    uint4 q[2];
};

// ---------------------------------------------------------------------------
// Elementwise f32 -> bf16 convert
// ---------------------------------------------------------------------------
__global__ __launch_bounds__(256)
void cvt_bf16_kernel(const float* __restrict__ in, bf16* __restrict__ out, int n)
{
    int i = blockIdx.x * 256 + threadIdx.x;
    if (i < n) out[i] = (bf16)in[i];
}

// ---------------------------------------------------------------------------
// Weight prep: f32 [K,N] -> bf16 [N,K] (transposed, so WMMA B-fragments are
// two contiguous 16-byte loads per lane).  grid = N, block = 256.
// ---------------------------------------------------------------------------
__global__ __launch_bounds__(256)
void twt_kernel(const float* __restrict__ in, bf16* __restrict__ out, int K, int N)
{
    int n = blockIdx.x;
    for (int k = threadIdx.x; k < K; k += 256)
        out[(size_t)n * K + k] = (bf16)in[(size_t)k * N + n];
}

// ---------------------------------------------------------------------------
// Cost transpose: [B,N,N] -> [B,N,N]^T per batch.  grid = B, block = 256.
// ---------------------------------------------------------------------------
__global__ __launch_bounds__(256)
void tcost_kernel(const float* __restrict__ in, float* __restrict__ out)
{
    int b = blockIdx.x;
    int j = threadIdx.x;
    for (int i = 0; i < N_; i++)
        out[((size_t)b * N_ + j) * N_ + i] = in[((size_t)b * N_ + i) * N_ + j];
}

// ---------------------------------------------------------------------------
// WMMA GEMM:  C[M,N] = A[M,K](bf16) x W[K,N]  with W given transposed [N,K].
// Fused epilogue: +bias[N], +addsrc[M,N] (residual / partial-sum), ReLU,
// stores f32 and/or bf16.  One wave (32 threads) per 64x64 output tile.
// ---------------------------------------------------------------------------
__global__ __launch_bounds__(32)
void gemm_bf16_wmma(const bf16* __restrict__ A,      // [M,K] row-major
                    const bf16* __restrict__ WT,     // [N,K] row-major (= W^T)
                    const float* __restrict__ bias,  // [N]   or nullptr
                    const float* __restrict__ addsrc,// [M,N] or nullptr
                    float* __restrict__ outF,        // [M,N] or nullptr
                    bf16*  __restrict__ outB,        // [M,N] or nullptr
                    int M, int N, int K, int relu)
{
    const int lane = threadIdx.x;
    const int ln15 = lane & 15;
    const int hs   = lane >> 4;                 // half-select (lanes 0-15 / 16-31)
    const int tm   = blockIdx.x * 64;
    const int tn   = blockIdx.y * 64;
    (void)M;

    v8f acc[4][4];
    #pragma unroll
    for (int mi = 0; mi < 4; mi++)
        #pragma unroll
        for (int ni = 0; ni < 4; ni++)
            #pragma unroll
            for (int r = 0; r < 8; r++) acc[mi][ni][r] = 0.0f;

    for (int k0 = 0; k0 < K; k0 += 32) {
        Frag16 a[4], b[4];
        // A fragment: lane = M row; elems 0..7 <- K = k0+8*hs.., 8..15 <- K = k0+16+8*hs..
        #pragma unroll
        for (int mi = 0; mi < 4; mi++) {
            const bf16* ap = A + (size_t)(tm + mi * 16 + ln15) * K + k0 + 8 * hs;
            a[mi].q[0] = *(const uint4*)(ap);
            a[mi].q[1] = *(const uint4*)(ap + 16);
        }
        // B fragment: lane = N col; elems 0..7 <- K = k0+8*hs.., 8..15 <- K = k0+16+8*hs.. (W^T)
        #pragma unroll
        for (int ni = 0; ni < 4; ni++) {
            const bf16* bp = WT + (size_t)(tn + ni * 16 + ln15) * K + k0 + 8 * hs;
            b[ni].q[0] = *(const uint4*)(bp);
            b[ni].q[1] = *(const uint4*)(bp + 16);
        }
        #pragma unroll
        for (int mi = 0; mi < 4; mi++)
            #pragma unroll
            for (int ni = 0; ni < 4; ni++)
                acc[mi][ni] = __builtin_amdgcn_wmma_f32_16x16x32_bf16(
                    false, a[mi].v, false, b[ni].v, (short)0, acc[mi][ni],
                    false, false);
    }

    // Epilogue.  C layout: VGPR r, lanes 0-15 -> M=r, lanes 16-31 -> M=r+8; N = lane&15.
    #pragma unroll
    for (int mi = 0; mi < 4; mi++) {
        #pragma unroll
        for (int ni = 0; ni < 4; ni++) {
            #pragma unroll
            for (int r = 0; r < 8; r++) {
                int m = tm + mi * 16 + r + 8 * hs;
                int n = tn + ni * 16 + ln15;
                float v = acc[mi][ni][r];
                if (bias)   v += bias[n];
                if (addsrc) v += addsrc[(size_t)m * N + n];
                if (relu)   v = v > 0.0f ? v : 0.0f;
                size_t idx = (size_t)m * N + n;
                if (outF) outF[idx] = v;
                if (outB) outB[idx] = (bf16)v;
            }
        }
    }
}

// ---------------------------------------------------------------------------
// Fused attention:  scores = (q k^T)/4 + cost ; softmax ; out = att @ v.
// One wave per (b, h, 16-query-row tile).  Score row tiles live entirely in
// VGPRs (16 x v8f); att and v staged in LDS (bf16) for the att@v WMMAs.
// DQK=16 is zero-padded to K=32 for the score WMMAs.
// grid = (16, H, B), block = 32.
// ---------------------------------------------------------------------------
__global__ __launch_bounds__(32)
void attn_wmma(const bf16* __restrict__ Qb,     // [B*N, H*DQK]
               const bf16* __restrict__ Kb,     // [B*N, H*DQK]  (k_s + k_a)
               const bf16* __restrict__ Vb,     // [B*N, H*DQK]
               const float* __restrict__ cost,  // [B, N, N]
               bf16* __restrict__ AOb)          // [B*N, H*DQK]
{
    __shared__ alignas(16) bf16 attL[16 * 256];   // 16 query rows x 256 keys
    __shared__ alignas(16) bf16 vL[256 * 16];     // v for this head

    const int lane = threadIdx.x;
    const int ln15 = lane & 15;
    const int hs   = lane >> 4;
    const int it   = blockIdx.x;   // query tile
    const int h    = blockIdx.y;
    const int b    = blockIdx.z;
    const float scale = 0.25f;     // 1/sqrt(DQK)

    // Stage v[:,h] (256 x 16 bf16) into LDS, coalesced 32B per lane.
    for (int j = lane; j < 256; j += 32) {
        const uint4* src = (const uint4*)(Vb + ((size_t)(b * 256 + j)) * (H_ * DQK_) + h * DQK_);
        uint4* dst = (uint4*)(vL + j * 16);
        dst[0] = src[0];
        dst[1] = src[1];
    }

    // q A-fragment, K padded 16 -> 32 (upper half zero).
    Frag16 qf;
    {
        const bf16* qp = Qb + ((size_t)(b * 256 + it * 16 + ln15)) * (H_ * DQK_) + h * DQK_ + 8 * hs;
        qf.q[0] = *(const uint4*)qp;
        qf.q[1] = uint4{0u, 0u, 0u, 0u};
    }

    // Scores: 16 key tiles, one WMMA each -> 16 x v8f in registers.
    v8f sc[16];
    #pragma unroll
    for (int jt = 0; jt < 16; jt++) {
        Frag16 kf;
        {   // elems 0..7 carry K(dqk)=8*hs..8*hs+7; elems 8..15 carry the zero pad
            const bf16* kp = Kb + ((size_t)(b * 256 + jt * 16 + ln15)) * (H_ * DQK_) + h * DQK_ + 8 * hs;
            kf.q[0] = *(const uint4*)kp;
            kf.q[1] = uint4{0u, 0u, 0u, 0u};
        }
        v8f z;
        #pragma unroll
        for (int r = 0; r < 8; r++) z[r] = 0.0f;
        sc[jt] = __builtin_amdgcn_wmma_f32_16x16x32_bf16(
            false, qf.v, false, kf.v, (short)0, z, false, false);
        #pragma unroll
        for (int r = 0; r < 8; r++) {
            int mrow = it * 16 + r + 8 * hs;
            int ncol = jt * 16 + ln15;
            sc[jt][r] = sc[jt][r] * scale + cost[((size_t)b * N_ + mrow) * N_ + ncol];
        }
    }

    // Row-wise softmax.  Each row's 256 values sit in one 16-lane half across
    // 16 tiles -> local reduce over tiles + shfl_xor over the 16-lane group
    // (masks < 16 never cross the half-wave boundary).
    #pragma unroll
    for (int r = 0; r < 8; r++) {
        float mx = -3.0e38f;
        #pragma unroll
        for (int jt = 0; jt < 16; jt++) mx = fmaxf(mx, sc[jt][r]);
        #pragma unroll
        for (int s = 1; s < 16; s <<= 1) mx = fmaxf(mx, __shfl_xor(mx, s, 32));
        float sum = 0.0f;
        #pragma unroll
        for (int jt = 0; jt < 16; jt++) {
            float p = __expf(sc[jt][r] - mx);
            sc[jt][r] = p;
            sum += p;
        }
        #pragma unroll
        for (int s = 1; s < 16; s <<= 1) sum += __shfl_xor(sum, s, 32);
        float inv = 1.0f / sum;
        int m = r + 8 * hs;
        #pragma unroll
        for (int jt = 0; jt < 16; jt++)
            attL[m * 256 + jt * 16 + ln15] = (bf16)(sc[jt][r] * inv);
    }
    __syncthreads();

    // out = att @ v : accumulate 8 K-chunks of 32.
    v8f oacc;
    #pragma unroll
    for (int r = 0; r < 8; r++) oacc[r] = 0.0f;
    #pragma unroll
    for (int c = 0; c < 8; c++) {
        Frag16 af, bfg;
        const bf16* ap = attL + ln15 * 256 + c * 32 + 8 * hs;
        af.q[0] = *(const uint4*)(ap);
        af.q[1] = *(const uint4*)(ap + 16);
        union { v16bf v; bf16 e[16]; } bb;
        #pragma unroll
        for (int e = 0; e < 16; e++)
            bb.e[e] = vL[(c * 32 + 8 * hs + (e & 7) + 16 * (e >> 3)) * 16 + ln15];
        bfg.v = bb.v;
        oacc = __builtin_amdgcn_wmma_f32_16x16x32_bf16(
            false, af.v, false, bfg.v, (short)0, oacc, false, false);
    }

    #pragma unroll
    for (int r = 0; r < 8; r++) {
        size_t row = (size_t)b * 256 + it * 16 + r + 8 * hs;
        AOb[row * (H_ * DQK_) + h * DQK_ + ln15] = (bf16)oacc[r];
    }
}

// ---------------------------------------------------------------------------
// InstanceNorm over the node axis per (batch, channel), affine per channel.
// Input is already residual-summed.  grid = B, block = 256 (thread = channel,
// fully coalesced).  Writes f32 + bf16 copies.
// ---------------------------------------------------------------------------
__global__ __launch_bounds__(256)
void inorm_kernel(const float* __restrict__ X,
                  const float* __restrict__ g, const float* __restrict__ be,
                  float* __restrict__ outF, bf16* __restrict__ outB)
{
    int b = blockIdx.x, c = threadIdx.x;
    const float* x = X + (size_t)b * N_ * EMB_ + c;
    float s = 0.0f, s2 = 0.0f;
    for (int n = 0; n < N_; n++) {
        float v = x[(size_t)n * EMB_];
        s += v;
        s2 += v * v;
    }
    float mean = s * (1.0f / N_);
    float var  = s2 * (1.0f / N_) - mean * mean;
    float rs   = rsqrtf(var + 1e-5f);
    float gg = g[c], bb = be[c];
    for (int n = 0; n < N_; n++) {
        float v = (x[(size_t)n * EMB_] - mean) * rs * gg + bb;
        size_t idx = (size_t)b * N_ * EMB_ + (size_t)n * EMB_ + c;
        outF[idx] = v;
        outB[idx] = (bf16)v;
    }
}

// ---------------------------------------------------------------------------
// Host orchestration
// ---------------------------------------------------------------------------
extern "C" void kernel_launch(void* const* d_in, const int* in_sizes, int n_in,
                              void* d_out, int out_size, void* d_ws, size_t ws_size,
                              hipStream_t stream)
{
    (void)in_sizes; (void)n_in; (void)out_size; (void)ws_size;

    const float* row_emb = (const float*)d_in[0];
    const float* col_emb = (const float*)d_in[1];
    const float* cost    = (const float*)d_in[2];
    const float* Wq  = (const float*)d_in[3];
    const float* Ws  = (const float*)d_in[4];
    const float* Wa  = (const float*)d_in[5];
    const float* Wv  = (const float*)d_in[6];
    const float* Wc  = (const float*)d_in[7];
    const float* bc  = (const float*)d_in[8];
    const float* g1  = (const float*)d_in[9];
    const float* be1 = (const float*)d_in[10];
    const float* W1  = (const float*)d_in[11];
    const float* bf1 = (const float*)d_in[12];
    const float* W2  = (const float*)d_in[13];
    const float* bf2 = (const float*)d_in[14];
    const float* g2  = (const float*)d_in[15];
    const float* be2 = (const float*)d_in[16];

    // ---- workspace carving (256B aligned) ----
    char* wptr = (char*)d_ws;
    auto carve = [&](size_t bytes) -> void* {
        void* p = (void*)wptr;
        wptr += (bytes + 255) & ~(size_t)255;
        return p;
    };
    const size_t S  = (size_t)MTOT * EMB_;      // 8192*256 activation elems
    const size_t ES = (size_t)EMB_ * (H_ * DQK_); // 65536 per weight matrix
    const size_t E1 = (size_t)EMB_ * FF_;         // 131072 (W1/W2)

    float* actF[2][2];
    bf16*  actB[2][2];
    for (int i = 0; i < 2; i++)
        for (int j = 0; j < 2; j++) {
            actF[i][j] = (float*)carve(S * 4);
            actB[i][j] = (bf16*)carve(S * 2);
        }
    float* costT = (float*)carve((size_t)B_ * N_ * N_ * 4);
    bf16* Qb   = (bf16*)carve(S * 2);
    bf16* Kbf  = (bf16*)carve(S * 2);
    bf16* Vbf  = (bf16*)carve(S * 2);
    bf16* AObf = (bf16*)carve(S * 2);
    float* tmpF = (float*)carve(S * 4);
    float* S12  = (float*)carve(S * 4);
    float* O1F  = (float*)carve(S * 4);
    bf16*  O1B  = (bf16*)carve(S * 2);
    bf16*  HffB = (bf16*)carve((size_t)MTOT * FF_ * 2);
    bf16* WqT = (bf16*)carve((size_t)L_ * 2 * ES * 2);
    bf16* WsT = (bf16*)carve((size_t)L_ * 2 * ES * 2);
    bf16* WaT = (bf16*)carve((size_t)L_ * 2 * ES * 2);
    bf16* WvT = (bf16*)carve((size_t)L_ * 2 * ES * 2);
    bf16* WcT = (bf16*)carve((size_t)L_ * 2 * ES * 2);
    bf16* W1T = (bf16*)carve((size_t)L_ * 2 * E1 * 2);
    bf16* W2T = (bf16*)carve((size_t)L_ * 2 * E1 * 2);

    // ---- prep: activations, cost^T, transposed bf16 weights ----
    hipMemcpyAsync(actF[0][0], row_emb, S * 4, hipMemcpyDeviceToDevice, stream);
    hipMemcpyAsync(actF[0][1], col_emb, S * 4, hipMemcpyDeviceToDevice, stream);
    cvt_bf16_kernel<<<(int)(S / 256), 256, 0, stream>>>(row_emb, actB[0][0], (int)S);
    cvt_bf16_kernel<<<(int)(S / 256), 256, 0, stream>>>(col_emb, actB[0][1], (int)S);
    tcost_kernel<<<B_, 256, 0, stream>>>(cost, costT);
    for (int ls = 0; ls < L_ * 2; ls++) {
        twt_kernel<<<256, 256, 0, stream>>>(Wq + (size_t)ls * ES, WqT + (size_t)ls * ES, 256, 256);
        twt_kernel<<<256, 256, 0, stream>>>(Ws + (size_t)ls * ES, WsT + (size_t)ls * ES, 256, 256);
        twt_kernel<<<256, 256, 0, stream>>>(Wa + (size_t)ls * ES, WaT + (size_t)ls * ES, 256, 256);
        twt_kernel<<<256, 256, 0, stream>>>(Wv + (size_t)ls * ES, WvT + (size_t)ls * ES, 256, 256);
        twt_kernel<<<256, 256, 0, stream>>>(Wc + (size_t)ls * ES, WcT + (size_t)ls * ES, 256, 256);
        twt_kernel<<<512, 256, 0, stream>>>(W1 + (size_t)ls * E1, W1T + (size_t)ls * E1, 256, 512);
        twt_kernel<<<256, 256, 0, stream>>>(W2 + (size_t)ls * E1, W2T + (size_t)ls * E1, 512, 256);
    }

    auto gemm = [&](const bf16* A, const bf16* WTp, const float* bias,
                    const float* add, float* oF, bf16* oB, int N, int K, int relu) {
        dim3 g(MTOT / 64, N / 64);
        gemm_bf16_wmma<<<g, 32, 0, stream>>>(A, WTp, bias, add, oF, oB, MTOT, N, K, relu);
    };

    // ---- 5 layers x (row block, col block) ----
    int cur = 0;
    for (int l = 0; l < L_; l++) {
        for (int s = 0; s < 2; s++) {
            int ls = l * 2 + s;
            const bf16*  xrB = actB[cur][s];
            const bf16*  xcB = actB[cur][1 - s];
            const float* xrF = actF[cur][s];
            const float* cst = (s == 0) ? cost : costT;
            float* noF = actF[1 - cur][s];
            bf16*  noB = actB[1 - cur][s];

            // q = xr@Wq ; k = xr@Ws + xc@Wa ; v = xc@Wv   (bf16 outputs)
            gemm(xrB, WqT + (size_t)ls * ES, nullptr, nullptr, nullptr, Qb,  256, 256, 0);
            gemm(xrB, WsT + (size_t)ls * ES, nullptr, nullptr, tmpF,   nullptr, 256, 256, 0);
            gemm(xcB, WaT + (size_t)ls * ES, nullptr, tmpF,   nullptr, Kbf, 256, 256, 0);
            gemm(xcB, WvT + (size_t)ls * ES, nullptr, nullptr, nullptr, Vbf, 256, 256, 0);

            // fused attention (scores never touch HBM)
            attn_wmma<<<dim3(16, H_, B_), 32, 0, stream>>>(Qb, Kbf, Vbf, cst, AObf);

            // mh = AO@Wc + bc + residual(xr) ; instance-norm 1
            gemm(AObf, WcT + (size_t)ls * ES, bc + (size_t)ls * EMB_, xrF, S12, nullptr, 256, 256, 0);
            inorm_kernel<<<B_, 256, 0, stream>>>(S12, g1 + (size_t)ls * EMB_,
                                                 be1 + (size_t)ls * EMB_, O1F, O1B);

            // ff = relu(o1@W1 + bf1)@W2 + bf2 + residual(o1) ; instance-norm 2
            gemm(O1B, W1T + (size_t)ls * E1, bf1 + (size_t)ls * FF_, nullptr, nullptr, HffB, 512, 256, 1);
            gemm(HffB, W2T + (size_t)ls * E1, bf2 + (size_t)ls * EMB_, O1F, S12, nullptr, 256, 512, 0);
            inorm_kernel<<<B_, 256, 0, stream>>>(S12, g2 + (size_t)ls * EMB_,
                                                 be2 + (size_t)ls * EMB_, noF, noB);
        }
        cur = 1 - cur;
    }

    // ---- output: (row, col) concatenated flat ----
    hipMemcpyAsync(d_out, actF[cur][0], S * 4, hipMemcpyDeviceToDevice, stream);
    hipMemcpyAsync((float*)d_out + S, actF[cur][1], S * 4, hipMemcpyDeviceToDevice, stream);
}
